// PlainGCN3D_14353780703617
// MI455X (gfx1250) — matrix operand order repaired and checked by reading, not compiled, not run on hardware
//
#include <hip/hip_runtime.h>

#define NPTS      16384
#define BATCH_PTS 4096
#define KNN       16

typedef __attribute__((ext_vector_type(8)))  float   v8f;
typedef __attribute__((ext_vector_type(8)))  __bf16  v8bf;
typedef __attribute__((ext_vector_type(16))) __bf16  v16bf;

__device__ __forceinline__ float bf2f(unsigned short h) {
    unsigned int u = ((unsigned int)h) << 16;
    return __uint_as_float(u);
}
__device__ __forceinline__ unsigned short f2bf(float f) {
    unsigned int u = __float_as_uint(f);
    u += 0x7FFFu + ((u >> 16) & 1u);   // round-to-nearest-even
    return (unsigned short)(u >> 16);
}

// ---------------------------------------------------------------------------
// KNN: one thread per query node; block = 256 contiguous nodes (same batch).
// Candidate positions staged through LDS in tiles of 256. Top-16 kept sorted
// in registers (includes self, dist ~ 0, matching top_k(-d2) semantics).
// ---------------------------------------------------------------------------
__global__ __launch_bounds__(256) void knn_kernel(const float* __restrict__ coords,
                                                  int* __restrict__ knn_idx) {
    __shared__ float sp[256][3];
    const int tid = threadIdx.x;
    const int n = blockIdx.x * 256 + tid;
    const int batch = n >> 12;           // 4096 nodes per batch
    const int colbase = batch << 12;

    const float qx = coords[n * 4 + 1];
    const float qy = coords[n * 4 + 2];
    const float qz = coords[n * 4 + 3];

    float bd[KNN];
    int   bi[KNN];
#pragma unroll
    for (int q = 0; q < KNN; ++q) { bd[q] = 3.0e38f; bi[q] = n; }

    for (int t = 0; t < BATCH_PTS / 256; ++t) {
        const int j = colbase + t * 256 + tid;
        sp[tid][0] = coords[j * 4 + 1];
        sp[tid][1] = coords[j * 4 + 2];
        sp[tid][2] = coords[j * 4 + 3];
        __syncthreads();
        for (int c = 0; c < 256; ++c) {
            const float dx = qx - sp[c][0];
            const float dy = qy - sp[c][1];
            const float dz = qz - sp[c][2];
            const float d = dx * dx + dy * dy + dz * dz;
            if (d < bd[KNN - 1]) {
                const int cand = colbase + t * 256 + c;
#pragma unroll
                for (int q = KNN - 1; q >= 0; --q) {
                    if (q > 0 && d < bd[q - 1]) { bd[q] = bd[q - 1]; bi[q] = bi[q - 1]; }
                    else { bd[q] = d; bi[q] = cand; break; }
                }
            }
        }
        __syncthreads();
    }
#pragma unroll
    for (int q = 0; q < KNN; ++q) knn_idx[n * KNN + q] = bi[q];
}

// ---------------------------------------------------------------------------
// Convert f32 -> bf16 (features).
// ---------------------------------------------------------------------------
__global__ void cvt_bf16_kernel(const float* __restrict__ in,
                                unsigned short* __restrict__ out, int n) {
    int i = blockIdx.x * blockDim.x + threadIdx.x;
    if (i < n) out[i] = f2bf(in[i]);
}

// ---------------------------------------------------------------------------
// Pack W (2C x COUT, f32) into bf16 WMMA B-fragments.
// Fragment order: [ct][kc][lane][elem0..15], lane = {half=lane>>4, n=lane&15},
// elem j -> k = kc*32 + half*16 + j (contiguous k order, same as the A loads: lanes 0-15 hold K=0..15,
// lanes 16-31 hold K=16..31). Rows k >= 2C padded with zeros.
// ---------------------------------------------------------------------------
template <int CIN, int COUT>
__global__ void pack_w_kernel(const float* __restrict__ W,
                              unsigned short* __restrict__ pW) {
    constexpr int K2  = 2 * CIN;
    constexpr int K2P = (K2 < 32) ? 32 : K2;
    constexpr int CHUNKS = K2P / 32;
    constexpr int TILES  = COUT / 16;
    const int total = TILES * CHUNKS * 32 * 16;
    for (int i = blockIdx.x * blockDim.x + threadIdx.x; i < total;
         i += gridDim.x * blockDim.x) {
        const int j    = i & 15;
        const int lane = (i >> 4) & 31;
        const int tk   = i >> 9;               // ct*CHUNKS + kc
        const int kc   = tk % CHUNKS;
        const int ct   = tk / CHUNKS;
        const int half = lane >> 4;
        const int nn   = lane & 15;
        const int k    = kc * 32 + half * 16 + j;
        const float v  = (k < K2) ? W[k * COUT + ct * 16 + nn] : 0.0f;
        pW[i] = f2bf(v);
    }
}

// ---------------------------------------------------------------------------
// EdgeConv, register-blocked over NPB nodes per block:
//   - one wave per 16-column output tile (TILES = COUT/16 waves per block)
//   - each block builds NPB edge matrices (16 x 2C, bf16, K padded to mult 32)
//     in LDS; neighbor indices staged in LDS; gathers vectorized (v8bf) when
//     CIN % 8 == 0
//   - main loop: load ONE B fragment per k-chunk, apply it to NPB A matrices
//     (NPB WMMAs per 32B weight load -> NPB-fold weight-traffic reduction)
//   - epilogue: bias + relu, max over 16 M-rows (8 acc regs + shfl_xor 16)
// ---------------------------------------------------------------------------
union AFrag { v16bf v; v8bf h[2]; };

template <int CIN, int COUT, bool OUT_F32, int NPB>
__global__ __launch_bounds__(32 * (COUT / 16)) void edgeconv_kernel(
    const unsigned short* __restrict__ xin,  // N x CIN bf16
    const int* __restrict__ knn_idx,         // N x 16
    const unsigned short* __restrict__ pW,   // packed bf16 weights
    const float* __restrict__ bias,          // COUT
    unsigned short* __restrict__ xout_bf,    // N x COUT (if !OUT_F32)
    float* __restrict__ xout_f) {            // N x COUT (if OUT_F32)
    constexpr int K2     = 2 * CIN;
    constexpr int K2P    = (K2 < 32) ? 32 : K2;
    constexpr int CHUNKS = K2P / 32;
    constexpr int TILES  = COUT / 16;
    constexpr int NTHR   = 32 * TILES;

    __shared__ __align__(16) unsigned short e[NPB * 16 * K2P];
    __shared__ int sidx[NPB * 16];

    const int nbase = blockIdx.x * NPB;
    const int tid = threadIdx.x;

    // Stage neighbor indices once.
    if (tid < NPB * 16) sidx[tid] = knn_idx[nbase * KNN + tid];
    __syncthreads();

    // Build NPB edge matrices: e[nd][m][k] = k<CIN ? xi[k] : xj[m][k-CIN]-xi[k-CIN]
    if constexpr (CIN % 8 == 0) {
        constexpr int KG = K2P / 8;                       // 8-channel groups
        for (int i = tid; i < NPB * 16 * KG; i += NTHR) {
            const int kg = i % KG;
            const int m  = (i / KG) & 15;
            const int nd = i / (KG * 16);
            const int k  = kg * 8;
            v8bf val = {};
            if (k < CIN) {
                val = *(const v8bf*)(const void*)(xin + (size_t)(nbase + nd) * CIN + k);
            } else if (k < K2) {
                const int c = k - CIN;
                const int j = sidx[nd * 16 + m];
                const v8bf xi = *(const v8bf*)(const void*)(xin + (size_t)(nbase + nd) * CIN + c);
                const v8bf xj = *(const v8bf*)(const void*)(xin + (size_t)j * CIN + c);
#pragma unroll
                for (int q = 0; q < 8; ++q)
                    val[q] = (__bf16)((float)xj[q] - (float)xi[q]);
            }
            *(v8bf*)(void*)&e[(nd * 16 + m) * K2P + k] = val;
        }
    } else {
        for (int i = tid; i < NPB * 16 * K2P; i += NTHR) {
            const int k  = i % K2P;
            const int m  = (i / K2P) & 15;
            const int nd = i / (K2P * 16);
            float v = 0.0f;
            if (k < K2) {
                if (k < CIN) {
                    v = bf2f(xin[(nbase + nd) * CIN + k]);
                } else {
                    const int c = k - CIN;
                    const int j = sidx[nd * 16 + m];
                    v = bf2f(xin[j * CIN + c]) - bf2f(xin[(nbase + nd) * CIN + c]);
                }
            }
            e[(nd * 16 + m) * K2P + k] = f2bf(v);
        }
    }
    __syncthreads();

    const int wv   = tid >> 5;     // output column tile ct
    const int lane = tid & 31;
    const int half = lane >> 4;
    const int col  = lane & 15;

    v8f acc[NPB];
#pragma unroll
    for (int nd = 0; nd < NPB; ++nd) acc[nd] = (v8f){};

    for (int kc = 0; kc < CHUNKS; ++kc) {
        // One B fragment per k-chunk, reused for NPB nodes.
        const v16bf bfrag =
            *(const v16bf*)(const void*)&pW[((wv * CHUNKS + kc) * 32 + lane) * 16];
#pragma unroll
        for (int nd = 0; nd < NPB; ++nd) {
            const unsigned short* arow =
                &e[(nd * 16 + col) * K2P + kc * 32 + half * 16];
            AFrag a;
            a.h[0] = *(const v8bf*)(const void*)(arow);        // K = kc*32+half*16+0..7
            a.h[1] = *(const v8bf*)(const void*)(arow + 8);    // K = kc*32+half*16+8..15
            acc[nd] = __builtin_amdgcn_wmma_f32_16x16x32_bf16(
                false, a.v, false, bfrag, (short)0, acc[nd], false, false);
        }
    }

    // Bias + ReLU + max over 16 neighbor rows, per node.
    const float bv = bias[wv * 16 + col];
#pragma unroll
    for (int nd = 0; nd < NPB; ++nd) {
        float m = 0.0f;                  // relu floor: max_k relu(h) >= 0
#pragma unroll
        for (int r = 0; r < 8; ++r) {
            float v = acc[nd][r] + bv;
            v = v > 0.0f ? v : 0.0f;
            m = v > m ? v : m;
        }
        const float o = __shfl_xor(m, 16, 32);   // partner half-wave: M = r+8
        m = m > o ? m : o;
        if (lane < 16) {
            const size_t off = (size_t)(nbase + nd) * COUT + wv * 16 + col;
            if (OUT_F32) xout_f[off] = m;
            else         xout_bf[off] = f2bf(m);
        }
    }
}

// ---------------------------------------------------------------------------
// Launcher
// ---------------------------------------------------------------------------
extern "C" void kernel_launch(void* const* d_in, const int* in_sizes, int n_in,
                              void* d_out, int out_size, void* d_ws, size_t ws_size,
                              hipStream_t stream) {
    const float* coords = (const float*)d_in[0];
    const float* feats  = (const float*)d_in[1];
    const float* W1 = (const float*)d_in[2];  const float* b1 = (const float*)d_in[3];
    const float* W2 = (const float*)d_in[4];  const float* b2 = (const float*)d_in[5];
    const float* W3 = (const float*)d_in[6];  const float* b3 = (const float*)d_in[7];
    (void)in_sizes; (void)n_in; (void)out_size; (void)ws_size;

    char* ws = (char*)d_ws;
    size_t off = 0;
    auto walloc = [&](size_t bytes) -> void* {
        off = (off + 255) & ~(size_t)255;
        void* p = ws + off;
        off += bytes;
        return p;
    };

    int*            idx = (int*)           walloc((size_t)NPTS * KNN * 4);
    unsigned short* x0  = (unsigned short*)walloc((size_t)NPTS * 4   * 2);
    unsigned short* x1  = (unsigned short*)walloc((size_t)NPTS * 64  * 2);
    unsigned short* x2  = (unsigned short*)walloc((size_t)NPTS * 128 * 2);
    // packed weight sizes: TILES*CHUNKS*32*16 bf16 elems
    unsigned short* pW1 = (unsigned short*)walloc((size_t)(64 / 16)  * 1  * 512 * 2);
    unsigned short* pW2 = (unsigned short*)walloc((size_t)(128 / 16) * 4  * 512 * 2);
    unsigned short* pW3 = (unsigned short*)walloc((size_t)(256 / 16) * 16 * 512 * 2);

    // Stage 0: conversions / weight packing
    {
        const int n = NPTS * 4;
        cvt_bf16_kernel<<<(n + 255) / 256, 256, 0, stream>>>(feats, x0, n);
    }
    pack_w_kernel<4, 64><<<8, 256, 0, stream>>>(W1, pW1);
    pack_w_kernel<64, 128><<<64, 256, 0, stream>>>(W2, pW2);
    pack_w_kernel<128, 256><<<512, 256, 0, stream>>>(W3, pW3);

    // Stage 1: KNN (4 batches x 4096, batches are contiguous blocks of rows)
    knn_kernel<<<NPTS / 256, 256, 0, stream>>>(coords, idx);

    // Stage 2: three EdgeConv layers (WMMA bf16, f32 accum, NPB-node blocking)
    edgeconv_kernel<4, 64, false, 8><<<NPTS / 8, 32 * (64 / 16), 0, stream>>>(
        x0, idx, pW1, b1, x1, nullptr);
    edgeconv_kernel<64, 128, false, 8><<<NPTS / 8, 32 * (128 / 16), 0, stream>>>(
        x1, idx, pW2, b2, x2, nullptr);
    edgeconv_kernel<128, 256, true, 4><<<NPTS / 4, 32 * (256 / 16), 0, stream>>>(
        x2, idx, pW3, b3, nullptr, (float*)d_out);
}
